// HierarchicalFeaturesExtractor_38405597561793
// MI455X (gfx1250) — matrix operand order repaired and checked by reading, not compiled, not run on hardware
//
#include <hip/hip_runtime.h>

#define Bsz 64
#define Ssz 1024
#define Dsz 512

typedef __attribute__((ext_vector_type(16))) __bf16       bf16x16;
typedef __attribute__((ext_vector_type(8)))  float        f32x8;
typedef __attribute__((ext_vector_type(4)))  unsigned int u32x4;
typedef __attribute__((ext_vector_type(2)))  unsigned int u32x2;

union Frag { bf16x16 v; u32x4 q[2]; };

__device__ __forceinline__ unsigned short f32_to_bf16_rne(float x) {
  union { float f; unsigned u; } a; a.f = x;
  unsigned r = a.u + 0x7FFFu + ((a.u >> 16) & 1u);
  return (unsigned short)(r >> 16);
}
__device__ __forceinline__ float bf16_bits_to_f32(unsigned short h) {
  union { unsigned u; float f; } a; a.u = ((unsigned)h) << 16;
  return a.f;
}
__device__ __forceinline__ void split2(float x, unsigned short& hi, unsigned short& lo) {
  hi = f32_to_bf16_rne(x);
  lo = f32_to_bf16_rne(x - bf16_bits_to_f32(hi));
}

// ---------------------------------------------------------------------------
// Kernel 1: combine conv taps into 6 matrices (+ the single-tap w1 matrix),
// split f32 -> bf16 hi/lo, and pre-pack into WMMA B-fragment layout:
//   B is 32(K) x 16(N) bf16: lanes 0-15 hold K=0..7,16..23 (2 per VGPR), lanes 16-31
//   hold K=8..15,24..31; N = lane % 16.  Storage: [t][k32][ntile16][lane][8 dwords].
// ---------------------------------------------------------------------------
__global__ void prep_weights(const float* __restrict__ w1, const float* __restrict__ b1,
                             const float* __restrict__ w2, const float* __restrict__ b2,
                             const float* __restrict__ w3, const float* __restrict__ b3,
                             const float* __restrict__ w4, const float* __restrict__ b4,
                             unsigned int* __restrict__ p6_hi, unsigned int* __restrict__ p6_lo,
                             unsigned int* __restrict__ p1_hi, unsigned int* __restrict__ p1_lo,
                             float* __restrict__ bias6, float* __restrict__ bias1) {
  int gid = blockIdx.x * blockDim.x + threadIdx.x;
  if (gid < Dsz) {
    bias6[gid] = b1[gid] + b2[gid] + 2.0f * b3[gid] + 3.0f * b4[gid];
    bias1[gid] = b1[gid];
  }
  int t   = gid / (Dsz * Dsz);
  int rem = gid - t * (Dsz * Dsz);
  int k   = rem / Dsz;
  int n   = rem - k * Dsz;
  if (t >= 7) return;
#define W_(w, j) (w)[((j) * Dsz + k) * Dsz + n]
  float v;
  switch (t) {
    case 0: v = W_(w4,0); break;
    case 1: v = W_(w3,0) + W_(w4,0) + W_(w4,1); break;
    case 2: v = W_(w1,0) + W_(w2,0) + W_(w3,0) + W_(w3,1) + W_(w4,0) + W_(w4,1) + W_(w4,2); break;
    case 3: v = W_(w2,1) + W_(w3,1) + W_(w3,2) + W_(w4,1) + W_(w4,2) + W_(w4,3); break;
    case 4: v = W_(w3,2) + W_(w4,2) + W_(w4,3); break;
    case 5: v = W_(w4,3); break;
    default: v = W_(w1,0); break;   // t == 6 -> single-tap (w1) matrix
  }
#undef W_
  unsigned short hi, lo; split2(v, hi, lo);
  int kk = k & 31, nn = n & 15;
  int lane = nn + (((kk >> 3) & 1) << 4);
  int kp = (kk & 7) | ((kk >> 4) << 3);
  int vreg = kp >> 1, half = kp & 1;
  int tt = (t < 6) ? t : 0;
  size_t dw = ((((size_t)tt * 16 + (size_t)(k >> 5)) * 32 + (size_t)(n >> 4)) * 32 + lane) * 8 + vreg;
  unsigned short* dh = (unsigned short*)((t < 6) ? p6_hi : p1_hi);
  unsigned short* dl = (unsigned short*)((t < 6) ? p6_lo : p1_lo);
  dh[dw * 2 + half] = hi;
  dl[dw * 2 + half] = lo;
}

// ---------------------------------------------------------------------------
// Kernel 2: tap-fused conv as a WMMA GEMM.
// Block = 256 threads = 8 waves; tile = 64 positions x 128 features.
// B weights are staged into LDS with ASYNC copies (global_load_async_to_lds_b128,
// ASYNCcnt), removing the 4x redundant per-M-wave global B traffic; A comes
// from a gathered+split bf16 hi/lo LDS tile. Each A-fragment pair feeds
// 12 WMMAs (4 N-tiles x bf16x3 split precision).
// ---------------------------------------------------------------------------
template <int NTAPS, int TC>
__global__ __launch_bounds__(256)
void conv_wmma(const int* __restrict__ tokens, const float* __restrict__ emb,
               const unsigned int* __restrict__ pk,   // hi array; lo at +hl_stride bytes
               unsigned hl_stride,                     // bytes from hi array to lo array
               const float* __restrict__ bias, float* __restrict__ out, int p_base) {
  __shared__ unsigned short sh_hi[69 * 72];            // feats hi (K-chunk of 64, pad 8)
  __shared__ unsigned short sh_lo[69 * 72];            // feats lo
  __shared__ __align__(16) unsigned char sh_b[32768];  // staged B: hi[2kg][8KB], lo at +16KB
  __shared__ int            sh_tok[72];

  int bx = blockIdx.x;
  int b     = bx >> 5;
  int mt    = (bx >> 2) & 7;
  int nt128 = bx & 3;
  int p0  = p_base + mt * 64;
  int n0  = nt128 * 128;
  int n0t = nt128 * 8;                                 // first 16-wide N tile index

  int tid  = threadIdx.x;
  int lane = tid & 31;
  int wave = tid >> 5;
  int wm = wave & 3;   // M sub-tile (x16)
  int wn = wave >> 2;  // N sub-tile (x64)

  if (tid < 69) {
    int p = p0 + tid - 2;
    p = p < 0 ? 0 : (p > Ssz - 1 ? Ssz - 1 : p);
    sh_tok[tid] = tokens[b * Ssz + p];
  }
  __syncthreads();

  f32x8 c[4] = {};
  int g    = lane >> 4;
  int mrow = lane & 15;

  unsigned ldsb = (unsigned)(size_t)(void*)sh_b;
  unsigned long long gbase = (unsigned long long)(size_t)pk;

  for (int kb = 0; kb < 8; ++kb) {          // K chunks of 64
    // ---- stage feats chunk: float4 gather + hi/lo split + packed b64 stores
    for (int idx = tid; idx < 69 * 16; idx += 256) {
      int row = idx >> 4;
      int c4  = (idx & 15) << 2;
      int tok = sh_tok[row];
      float4 x = *(const float4*)(emb + (size_t)tok * Dsz + kb * 64 + c4);
      unsigned short h0, l0, h1, l1, h2, l2, h3, l3;
      split2(x.x, h0, l0); split2(x.y, h1, l1);
      split2(x.z, h2, l2); split2(x.w, h3, l3);
      u32x2 hp = { (unsigned)h0 | ((unsigned)h1 << 16), (unsigned)h2 | ((unsigned)h3 << 16) };
      u32x2 lp = { (unsigned)l0 | ((unsigned)l1 << 16), (unsigned)l2 | ((unsigned)l3 << 16) };
      *(u32x2*)(sh_hi + row * 72 + c4) = hp;
      *(u32x2*)(sh_lo + row * 72 + c4) = lp;
    }

    for (int t = 0; t < NTAPS; ++t) {
      // ---- async-stage B(t, this K-chunk, this N-slice) into LDS (32 KB)
      for (int idx = tid; idx < 2048; idx += 256) {
        int arr = idx >> 10;                 // 0 = hi, 1 = lo
        int rem = idx & 1023;
        int kgl = rem >> 9;                  // K32 sub-chunk 0..1
        int wi  = rem & 511;                 // b128 within 8KB chunk
        int kg  = kb * 2 + kgl;
        unsigned voff = (arr ? hl_stride : 0u)
                      + ((((unsigned)t * 16 + (unsigned)kg) * 32 + (unsigned)n0t) << 10)
                      + ((unsigned)wi << 4);
        unsigned lds = ldsb + (unsigned)((arr << 14) + (kgl << 13) + (wi << 4));
        asm volatile("global_load_async_to_lds_b128 %0, %1, %2"
                     :: "v"(lds), "v"(voff), "s"(gbase) : "memory");
      }
      asm volatile("s_wait_asynccnt 0x0" ::: "memory");
      __syncthreads();                       // B staged + (t==0) feats visible

      int rbase = wm * 16 + mrow + t + (2 - TC);   // staged feats row
#pragma unroll
      for (int kc = 0; kc < 2; ++kc) {             // K sub-chunks of 32
        int abase = rbase * 72 + kc * 32 + g * 8;
        Frag ah, al;
        ah.q[0] = *(const u32x4*)(sh_hi + abase);
        ah.q[1] = *(const u32x4*)(sh_hi + abase + 16);
        al.q[0] = *(const u32x4*)(sh_lo + abase);
        al.q[1] = *(const u32x4*)(sh_lo + abase + 16);
#pragma unroll
        for (int nt = 0; nt < 4; ++nt) {
          int ntl = wn * 4 + nt;
          const u32x4* ph = (const u32x4*)(sh_b + (kc << 13) + ((ntl * 32 + lane) << 5));
          const u32x4* pl = (const u32x4*)(sh_b + 16384 + (kc << 13) + ((ntl * 32 + lane) << 5));
          Frag bh, bl;
          bh.q[0] = ph[0]; bh.q[1] = ph[1];
          bl.q[0] = pl[0]; bl.q[1] = pl[1];
          c[nt] = __builtin_amdgcn_wmma_f32_16x16x32_bf16(false, ah.v, false, bh.v, (short)0, c[nt], false, false);
          c[nt] = __builtin_amdgcn_wmma_f32_16x16x32_bf16(false, ah.v, false, bl.v, (short)0, c[nt], false, false);
          c[nt] = __builtin_amdgcn_wmma_f32_16x16x32_bf16(false, al.v, false, bh.v, (short)0, c[nt], false, false);
        }
      }
      __syncthreads();                       // all waves done reading B / feats
    }
  }

  int nn = lane & 15;
#pragma unroll
  for (int nt = 0; nt < 4; ++nt) {
    int col = n0 + wn * 64 + nt * 16 + nn;
    float bv = bias[col];
#pragma unroll
    for (int r = 0; r < 8; ++r) {
      int row = p0 + wm * 16 + r + 8 * g;    // C layout: M = r (+8 for hi lanes)
      out[((size_t)b * Ssz + row) * Dsz + col] = c[nt][r] + bv;
    }
  }
}

// ---------------------------------------------------------------------------
// Kernel 3: exact f32 recompute of the edge rows i = 0, 1 of each batch.
// ---------------------------------------------------------------------------
__global__ void fixup_edges(const int* __restrict__ tokens, const float* __restrict__ emb,
                            const float* __restrict__ w1, const float* __restrict__ b1,
                            const float* __restrict__ w2, const float* __restrict__ b2,
                            const float* __restrict__ w3, const float* __restrict__ b3,
                            const float* __restrict__ w4, const float* __restrict__ b4,
                            float* __restrict__ out) {
  int b = blockIdx.x >> 1;
  int i = blockIdx.x & 1;
  __shared__ float f[5][Dsz];
  int tid = threadIdx.x;
  for (int idx = tid; idx < 5 * Dsz; idx += 256) {
    int r = idx / Dsz, k = idx - (idx / Dsz) * Dsz;
    int tok = tokens[b * Ssz + r];
    f[r][k] = emb[(size_t)tok * Dsz + k];
  }
  __syncthreads();
  for (int nb = 0; nb < 2; ++nb) {
    int n = tid + nb * 256;
    float acc = 0.0f;
    if (i == 0) {
      for (int k = 0; k < Dsz; ++k) {
        float c0 = w1[k * Dsz + n] + w2[k * Dsz + n] + w3[k * Dsz + n] + w4[k * Dsz + n];
        float c1 = w2[(Dsz + k) * Dsz + n] + w3[(Dsz + k) * Dsz + n] + w4[(Dsz + k) * Dsz + n];
        float c2 = w3[(2 * Dsz + k) * Dsz + n] + w4[(2 * Dsz + k) * Dsz + n];
        float c3 = w4[(3 * Dsz + k) * Dsz + n];
        acc += f[0][k] * c0 + f[1][k] * c1 + f[2][k] * c2 + f[3][k] * c3;
      }
      acc += b1[n] + b2[n] + b3[n] + b4[n];
    } else {
      for (int k = 0; k < Dsz; ++k) {
        float c0 = w3[k * Dsz + n] + w4[k * Dsz + n];
        float c1 = w1[k * Dsz + n] + w2[k * Dsz + n] + w3[k * Dsz + n] + w3[(Dsz + k) * Dsz + n]
                 + w4[k * Dsz + n] + w4[(Dsz + k) * Dsz + n];
        float c2 = w2[(Dsz + k) * Dsz + n] + w3[(Dsz + k) * Dsz + n] + w3[(2 * Dsz + k) * Dsz + n]
                 + w4[(Dsz + k) * Dsz + n] + w4[(2 * Dsz + k) * Dsz + n];
        float c3 = w3[(2 * Dsz + k) * Dsz + n] + w4[(2 * Dsz + k) * Dsz + n] + w4[(3 * Dsz + k) * Dsz + n];
        float c4 = w4[(3 * Dsz + k) * Dsz + n];
        acc += f[0][k] * c0 + f[1][k] * c1 + f[2][k] * c2 + f[3][k] * c3 + f[4][k] * c4;
      }
      acc += b1[n] + b2[n] + 2.0f * b3[n] + 2.0f * b4[n];
    }
    out[((size_t)b * Ssz + i) * Dsz + n] = acc;
  }
}

// ---------------------------------------------------------------------------
// Kernel 4: pad mask [B,S] (token==PAD) and causal mask [S,S] (triu, k=1).
// ---------------------------------------------------------------------------
__global__ void masks_kernel(const int* __restrict__ tokens,
                             float* __restrict__ pad_out, float* __restrict__ seq_out) {
  int idx = blockIdx.x * blockDim.x + threadIdx.x;
  if (idx < Bsz * Ssz) pad_out[idx] = (tokens[idx] == 0) ? 1.0f : 0.0f;
  if (idx < Ssz * Ssz) {
    int r = idx >> 10, cc = idx & 1023;
    seq_out[idx] = (cc > r) ? 1.0f : 0.0f;
  }
}

extern "C" void kernel_launch(void* const* d_in, const int* in_sizes, int n_in,
                              void* d_out, int out_size, void* d_ws, size_t ws_size,
                              hipStream_t stream) {
  (void)in_sizes; (void)n_in; (void)out_size; (void)ws_size;
  const int*   tokens = (const int*)d_in[0];
  const float* emb    = (const float*)d_in[1];
  const float* w1 = (const float*)d_in[2]; const float* b1 = (const float*)d_in[3];
  const float* w2 = (const float*)d_in[4]; const float* b2 = (const float*)d_in[5];
  const float* w3 = (const float*)d_in[6]; const float* b3 = (const float*)d_in[7];
  const float* w4 = (const float*)d_in[8]; const float* b4 = (const float*)d_in[9];
  float* out = (float*)d_out;

  unsigned int* ws = (unsigned int*)d_ws;
  const size_t P6 = (size_t)6 * 16 * 32 * 32 * 8;  // dwords
  const size_t P1 = (size_t)1 * 16 * 32 * 32 * 8;  // dwords
  unsigned int* p6_hi = ws;
  unsigned int* p6_lo = p6_hi + P6;                // contiguous: lo = hi + P6*4 bytes
  unsigned int* p1_hi = p6_lo + P6;
  unsigned int* p1_lo = p1_hi + P1;                // contiguous: lo = hi + P1*4 bytes
  float* bias6 = (float*)(p1_lo + P1);
  float* bias1 = bias6 + Dsz;

  // 1) combine + pack weights (7 * D * D threads)
  prep_weights<<<(7 * Dsz * Dsz) / 256, 256, 0, stream>>>(
      w1, b1, w2, b2, w3, b3, w4, b4, p6_hi, p6_lo, p1_hi, p1_lo, bias6, bias1);

  // 2) main conv GEMMs: 6-tap for i<512, 1-tap for i>=512  (64x128 tiles)
  conv_wmma<6, 2><<<Bsz * 8 * 4, 256, 0, stream>>>(
      tokens, emb, p6_hi, (unsigned)(P6 * 4), bias6, out, 0);
  conv_wmma<1, 0><<<Bsz * 8 * 4, 256, 0, stream>>>(
      tokens, emb, p1_hi, (unsigned)(P1 * 4), bias1, out, 512);

  // 3) exact edge rows i = 0, 1
  fixup_edges<<<Bsz * 2, 256, 0, stream>>>(tokens, emb, w1, b1, w2, b2, w3, b3, w4, b4, out);

  // 4) masks appended after uni
  float* pad_out = out + (size_t)Bsz * Ssz * Dsz;
  float* seq_out = pad_out + (size_t)Bsz * Ssz;
  masks_kernel<<<(Ssz * Ssz + 255) / 256, 256, 0, stream>>>(tokens, pad_out, seq_out);
}
